// Memory_3135326126764
// MI455X (gfx1250) — compile-verified, matrix operand order checked by reading, not run on hardware
//
#include <hip/hip_runtime.h>
#include <hip/hip_bf16.h>

// Problem dims (fixed by reference)
#define BIMG 8
#define CH   512
#define HH   96
#define WW   144
#define HW   (HH * WW)          // 13824
#define NTOK (BIMG * HW)        // 110592 tokens
#define DK   512                // feature dim == memory slots

typedef __attribute__((ext_vector_type(16))) __bf16 v16bf;
typedef __attribute__((ext_vector_type(8)))  float  v8f;

union FragBits { uint4 q[2]; v16bf v; };   // 32 bytes, 16-byte aligned

static __device__ __forceinline__ unsigned short f2bf(float f) {
    union { float f; unsigned int u; } x; x.f = f;
    unsigned int u = x.u;
    unsigned int r = (u + 0x7FFFu + ((u >> 16) & 1u)) >> 16;   // round-to-nearest-even
    return (unsigned short)r;
}

// ---------------------------------------------------------------------------
// K0: memory -> bf16 (row-major and transposed), zero column-sum accumulator
// ---------------------------------------------------------------------------
__global__ __launch_bounds__(256) void kprep(const float* __restrict__ mem,
                                             unsigned short* __restrict__ mem_bf,
                                             unsigned short* __restrict__ memT_bf,
                                             float* __restrict__ colsum) {
    int e = blockIdx.x * 256 + threadIdx.x;      // < 512*512
    int m = e >> 9, d = e & 511;
    unsigned short h = f2bf(mem[e]);
    mem_bf[e] = h;                               // B for GEMM2: B[k][n] = mem[k][n]
    memT_bf[d * 512 + m] = h;                    // B for GEMM1: B[k][n] = mem[n][k]
    if (e < 512) colsum[e] = 0.0f;
}

// ---------------------------------------------------------------------------
// K1: L2-normalize over channels; write q as bf16 [NTOK, 512] row-major.
// ---------------------------------------------------------------------------
__global__ __launch_bounds__(256) void knorm(const float* __restrict__ qs,
                                             unsigned short* __restrict__ qa) {
    int p = blockIdx.x * 256 + threadIdx.x;      // < NTOK
    int b = p / HW, hw = p - b * HW;
    const float* src = qs + (long)b * CH * HW + hw;
    float ss = 0.0f;
    for (int c = 0; c < CH; ++c) { float v = src[(long)c * HW]; ss += v * v; }
    float inv = 1.0f / fmaxf(sqrtf(ss), 1e-12f);
    unsigned short* dst = qa + (long)p * DK;
    for (int c = 0; c < CH; ++c) dst[c] = f2bf(src[(long)c * HW] * inv);
}

// ---------------------------------------------------------------------------
// GEMM via v_wmma_f32_16x16x32_bf16, 32x32 register tile per wave.
//   Block = 8 waves as 4(row) x 2(col): block tile 128 rows x 64 cols.
//   A: [NTOK,512] bf16 row-major; fragment = two global b128 loads (the 16
//      halfs a lane needs are two runs of 8 consecutive K values).
//   B: 512x64 slab staged TRANSPOSED in LDS, row pitch 520 halfs:
//      - fragment addr = colBase + kt*64 bytes -> pure immediate-offset
//        ds_load_b128, no per-iteration address VALU
//      - col stride 1040 B = 260 dwords = 4 banks -> 16 lanes x 4-bank b128
//        reads cover all 64 banks, conflict-free
//   Per kt: 4 global b128 + 4 ds b128 feed 4 WMMAs (1 vmem + 1 ds per WMMA).
//   MODE 0: Out[n*512+m] = exp(S - 1)          (scores bounded in [-1,1])
//   MODE 1: Out[((b*512+d)*HW)+hw] = S         (permuted [B,D,H,W] store)
// ---------------------------------------------------------------------------
#define BPITCH 520
template <int MODE>
__global__ __launch_bounds__(256) void gemm_wmma(const unsigned short* __restrict__ Abf,
                                                 const unsigned short* __restrict__ Brow,
                                                 float* __restrict__ Out) {
    __shared__ unsigned short bsT[64 * BPITCH];  // 66,560 B (CDNA5: up to 320 KB/WG)
    const int tid   = threadIdx.x;
    const int wave  = tid >> 5;
    const int lane  = tid & 31;
    const int col16 = lane & 15;
    const int half  = lane >> 4;
    const int m0    = blockIdx.y * 64;

    { // stage B slab transposed: bsT[col*BPITCH + k] = Brow[k*512 + m0 + col]
      // K-quad granularity: 4 coalesced u32 reads -> 2 packed ds_store_b64
        const unsigned int* Bu = (const unsigned int*)Brow;
        const int mbase = m0 >> 1;
        #pragma unroll
        for (int j = 0; j < 16; ++j) {
            int w  = tid + j * 256;              // 4096 items: 128 k-quads x 32 col-pairs
            int kq = w >> 5;
            int pr = w & 31;
            int k  = kq * 4;
            unsigned int v0 = Bu[(k + 0) * 256 + mbase + pr];
            unsigned int v1 = Bu[(k + 1) * 256 + mbase + pr];
            unsigned int v2 = Bu[(k + 2) * 256 + mbase + pr];
            unsigned int v3 = Bu[(k + 3) * 256 + mbase + pr];
            int c0 = 2 * pr, c1 = c0 + 1;
            uint2 lo, hi;
            lo.x = (v0 & 0xFFFFu) | (v1 << 16);
            lo.y = (v2 & 0xFFFFu) | (v3 << 16);
            hi.x = (v0 >> 16) | (v1 & 0xFFFF0000u);
            hi.y = (v2 >> 16) | (v3 & 0xFFFF0000u);
            *(uint2*)&bsT[c0 * BPITCH + k] = lo;
            *(uint2*)&bsT[c1 * BPITCH + k] = hi;
        }
    }
    __syncthreads();

    const int wr = wave & 3;                     // row-tile group (32 rows each)
    const int wc = wave >> 2;                    // col-tile group (32 cols each)
    const int row0 = blockIdx.x * 128 + wr * 32;

    // Lane's two A row bases (row-tile 0 and +16), offset by K-half subset
    const char* arow0 = (const char*)Abf + (long)(row0 + col16) * (DK * 2) + half * 16;
    const char* arow1 = arow0 + 16 * (DK * 2);
    // Lane's two B fragment bases in LDS (immediate offsets kt*64 / +32 from here)
    const unsigned short* bb0 = bsT + (wc * 32 + col16) * BPITCH + half * 8;
    const unsigned short* bb1 = bb0 + 16 * BPITCH;

    v8f c00 = {}, c01 = {}, c10 = {}, c11 = {};
    #pragma unroll
    for (int kt = 0; kt < 16; ++kt) {
        FragBits a0, a1, b0, b1;
        a0.q[0] = *(const uint4*)(arow0 + kt * 64);        // K = kt*32 + half*8 ..
        a0.q[1] = *(const uint4*)(arow0 + kt * 64 + 32);   // K = kt*32+16+half*8 ..
        a1.q[0] = *(const uint4*)(arow1 + kt * 64);
        a1.q[1] = *(const uint4*)(arow1 + kt * 64 + 32);
        b0.q[0] = *(const uint4*)(bb0 + kt * 32);
        b0.q[1] = *(const uint4*)(bb0 + kt * 32 + 16);
        b1.q[0] = *(const uint4*)(bb1 + kt * 32);
        b1.q[1] = *(const uint4*)(bb1 + kt * 32 + 16);
        c00 = __builtin_amdgcn_wmma_f32_16x16x32_bf16(false, a0.v, false, b0.v,
                                                      (short)0, c00, false, false);
        c01 = __builtin_amdgcn_wmma_f32_16x16x32_bf16(false, a0.v, false, b1.v,
                                                      (short)0, c01, false, false);
        c10 = __builtin_amdgcn_wmma_f32_16x16x32_bf16(false, a1.v, false, b0.v,
                                                      (short)0, c10, false, false);
        c11 = __builtin_amdgcn_wmma_f32_16x16x32_bf16(false, a1.v, false, b1.v,
                                                      (short)0, c11, false, false);
    }

    const int gc0 = m0 + wc * 32 + col16;        // global output columns
    const int gc1 = gc0 + 16;
    if (MODE == 0) {
        #pragma unroll
        for (int r = 0; r < 8; ++r) {
            int ra = row0 + r + half * 8;        // C/D layout: VGPR r -> M = r + 8*half
            int rb = ra + 16;
            Out[(long)ra * 512 + gc0] = __expf(c00[r] - 1.0f);
            Out[(long)ra * 512 + gc1] = __expf(c01[r] - 1.0f);
            Out[(long)rb * 512 + gc0] = __expf(c10[r] - 1.0f);
            Out[(long)rb * 512 + gc1] = __expf(c11[r] - 1.0f);
        }
    } else {
        #pragma unroll
        for (int r = 0; r < 8; ++r) {
            int na = row0 + r + half * 8;
            int nb = na + 16;
            int ba = na / HW, hwa = na - ba * HW;
            int bb = nb / HW, hwb = nb - bb * HW;
            Out[((long)ba * 512 + gc0) * HW + hwa] = c00[r];
            Out[((long)ba * 512 + gc1) * HW + hwa] = c01[r];
            Out[((long)bb * 512 + gc0) * HW + hwb] = c10[r];
            Out[((long)bb * 512 + gc1) * HW + hwb] = c11[r];
        }
    }
}

// ---------------------------------------------------------------------------
// K3: per-row sums of p = exp(S-1); store reciprocal. One wave32 per row.
// ---------------------------------------------------------------------------
__global__ __launch_bounds__(256) void krowsum(const float* __restrict__ p,
                                               float* __restrict__ rowinv) {
    int wave = threadIdx.x >> 5, lane = threadIdx.x & 31;
    int n = blockIdx.x * 8 + wave;
    const float* row = p + (long)n * 512;
    float s = 0.0f;
    #pragma unroll
    for (int i = 0; i < 16; ++i) s += row[lane + i * 32];
    #pragma unroll
    for (int off = 16; off > 0; off >>= 1) s += __shfl_xor(s, off, 32);
    if (lane == 0) rowinv[n] = 1.0f / s;
}

// ---------------------------------------------------------------------------
// K4: per-column partial sums (coalesced: thread = column), atomic fold.
// ---------------------------------------------------------------------------
__global__ __launch_bounds__(256) void kcolsum(const float* __restrict__ p,
                                               float* __restrict__ colsum) {
    const int t = threadIdx.x;
    const long r0 = (long)blockIdx.x * (NTOK / 512);  // 216 rows per block
    float a0 = 0.0f, a1 = 0.0f;
    for (int i = 0; i < NTOK / 512; ++i) {
        const float* row = p + (r0 + i) * 512;
        a0 += row[t];
        a1 += row[t + 256];
    }
    atomicAdd(&colsum[t], a0);
    atomicAdd(&colsum[t + 256], a1);
}

// ---------------------------------------------------------------------------
// K5: finalize both softmaxes + emit bf16 A-matrix for GEMM2. 4 elems/thread.
// ---------------------------------------------------------------------------
__global__ __launch_bounds__(256) void kfinal(float* __restrict__ p_inout,
                                              float* __restrict__ out1,
                                              const float* __restrict__ rowinv,
                                              const float* __restrict__ colsum,
                                              unsigned short* __restrict__ abf) {
    long base = ((long)blockIdx.x * 256 + threadIdx.x) * 4;
    int n = (int)(base >> 9);
    int m = (int)(base & 511);
    float4 pv = *(const float4*)(p_inout + base);
    float rv = rowinv[n];
    float4 o1, o2;
    o1.x = pv.x / colsum[m];     o1.y = pv.y / colsum[m + 1];
    o1.z = pv.z / colsum[m + 2]; o1.w = pv.w / colsum[m + 3];
    o2.x = pv.x * rv; o2.y = pv.y * rv; o2.z = pv.z * rv; o2.w = pv.w * rv;
    *(float4*)(out1 + base)    = o1;
    *(float4*)(p_inout + base) = o2;
    unsigned long long pack =
        (unsigned long long)f2bf(o2.x)
      | ((unsigned long long)f2bf(o2.y) << 16)
      | ((unsigned long long)f2bf(o2.z) << 32)
      | ((unsigned long long)f2bf(o2.w) << 48);
    *(unsigned long long*)(abf + base) = pack;
}

// ---------------------------------------------------------------------------
extern "C" void kernel_launch(void* const* d_in, const int* in_sizes, int n_in,
                              void* d_out, int out_size, void* d_ws, size_t ws_size,
                              hipStream_t stream) {
    const float* qs  = (const float*)d_in[0];   // [8,512,96,144]
    const float* mem = (const float*)d_in[1];   // [512,512]

    float* out  = (float*)d_out;
    float* out0 = out;                                    // updated_query  [B,D,H,W]
    float* out1 = out0 + (long)NTOK * 512;                // softmax_score_query  [N,M]
    float* out2 = out1 + (long)NTOK * 512;                // softmax_score_memory [N,M]

    char* ws = (char*)d_ws;
    unsigned short* qa      = (unsigned short*)ws;                       // 113,246,208 B
    unsigned short* mem_bf  = (unsigned short*)(ws + 113246208);         //     524,288 B
    unsigned short* memT_bf = (unsigned short*)(ws + 113770496);         //     524,288 B
    float*          rowinv  = (float*)(ws + 114294784);                  //     442,368 B
    float*          colsum  = (float*)(ws + 114737152);                  //       2,048 B

    kprep<<<1024, 256, 0, stream>>>(mem, mem_bf, memT_bf, colsum);
    knorm<<<NTOK / 256, 256, 0, stream>>>(qs, qa);

    // GEMM1: S = q @ mem^T ; store p = exp(S-1) into out2 region
    gemm_wmma<0><<<dim3(NTOK / 128, 512 / 64), 256, 0, stream>>>(qa, memT_bf, out2);

    krowsum<<<NTOK / 8, 256, 0, stream>>>(out2, rowinv);
    kcolsum<<<512, 256, 0, stream>>>(out2, colsum);

    // Finalize softmaxes; qa region is reused as bf16 A-matrix for GEMM2
    kfinal<<<(long)NTOK * 512 / 1024, 256, 0, stream>>>(out2, out1, rowinv, colsum, qa);

    // GEMM2: upd = softmax_row @ mem, written permuted to [B,D,H,W]
    gemm_wmma<1><<<dim3(NTOK / 128, 512 / 64), 256, 0, stream>>>(qa, mem_bf, out0);
}
